// EXL2Linear_1855425872553
// MI455X (gfx1250) — matrix operand order repaired and checked by reading, not compiled, not run on hardware
//
#include <hip/hip_runtime.h>

// y = x @ (q*s+z)^T + b ; M=8192, K=4096, N=11008.
// Compute-bound f16 WMMA GEMM with CDNA5 async global->LDS double buffering.
// Pass 1/2 pre-convert A (f32->f16) and W (int4-levels -> dequant f16) into d_ws.
// L2 policy: one-shot streams (x, qweight, out) use non-temporal hints so the
// 192MB L2 keeps the re-used f16 A/W operands resident.

typedef __attribute__((ext_vector_type(16))) _Float16 v16h;
typedef __attribute__((ext_vector_type(8)))  _Float16 v8h;
typedef __attribute__((ext_vector_type(4)))  _Float16 v4h;
typedef __attribute__((ext_vector_type(8)))  float    v8f;
typedef __attribute__((ext_vector_type(4)))  float    v4f;
typedef __attribute__((ext_vector_type(4)))  int      v4i;

static constexpr int K_DIM = 4096;
static constexpr int N_DIM = 11008;
static constexpr int M_DIM = 8192;

static constexpr int BM = 128;
static constexpr int BN = 128;
static constexpr int BK = 64;                 // two 16x16x32 k-substeps
static constexpr int NK = K_DIM / BK;         // 64 iterations
static constexpr int LDKH = BK + 8;           // 72 halves = 144 B stride (16B aligned,
                                              // 36-bank step -> conflict-free b128 frags)

// ---------------- pass 1: x f32 -> f16 ----------------
__global__ __launch_bounds__(256)
void cvt_x_f16(const float* __restrict__ x, _Float16* __restrict__ a16) {
    size_t i = ((size_t)blockIdx.x * 256 + threadIdx.x) * 4;
    const v4f f = __builtin_nontemporal_load((const v4f*)(x + i));   // read-once stream
    v4h h;
    h.x = (_Float16)f.x; h.y = (_Float16)f.y;
    h.z = (_Float16)f.z; h.w = (_Float16)f.w;
    *(v4h*)(a16 + i) = h;                     // re-read 86x by GEMM: keep cached
}

// ---------------- pass 2: dequantize W = q*s+z -> f16 ----------------
__global__ __launch_bounds__(256)
void dequant_w_f16(const int* __restrict__ qw,
                   const float* __restrict__ scales,
                   const float* __restrict__ zeros,
                   _Float16* __restrict__ w16) {
    const int o = blockIdx.y;
    const float sv = scales[o];
    const float zv = zeros[o];
    const size_t base = (size_t)o * K_DIM;
    const int col = (blockIdx.x * 256 + threadIdx.x) * 4;
    const v4i q = __builtin_nontemporal_load((const v4i*)(qw + base + col)); // read-once
    v4h h;
    h.x = (_Float16)fmaf((float)q.x, sv, zv);
    h.y = (_Float16)fmaf((float)q.y, sv, zv);
    h.z = (_Float16)fmaf((float)q.z, sv, zv);
    h.w = (_Float16)fmaf((float)q.w, sv, zv);
    *(v4h*)(w16 + base + col) = h;            // re-read 64x by GEMM: keep cached
}

// ---------------- CDNA5 async copy helpers ----------------
__device__ __forceinline__ void async_ld_b128(uint32_t lds_addr, uint64_t gaddr) {
    // global->LDS copy to the LDS address in vdst, 16 bytes per lane, tracked by ASYNCcnt
    asm volatile("global_load_async_to_lds_b128 %0, %1, off"
                 :: "v"(lds_addr), "v"(gaddr) : "memory");
}
__device__ __forceinline__ void wait_async_le8() {
    asm volatile("s_wait_asynccnt 0x8" ::: "memory");
}
__device__ __forceinline__ void wait_async_0() {
    asm volatile("s_wait_asynccnt 0x0" ::: "memory");
}

// ---------------- pass 3: GEMM ----------------
__global__ __launch_bounds__(256)
void exl2_wmma_gemm(const _Float16* __restrict__ A,   // [M,K] f16
                    const _Float16* __restrict__ W,   // [N,K] f16 (dequantized)
                    const float* __restrict__ bias,
                    float* __restrict__ out) {
    __shared__ _Float16 sA[2][BM * LDKH];
    __shared__ _Float16 sB[2][BN * LDKH];

    const int tid  = threadIdx.x;
    const int m0   = blockIdx.y * BM;
    const int n0   = blockIdx.x * BN;
    const int wave = tid >> 5;
    const int lane = tid & 31;
    const int wm   = wave & 3;        // 4 waves along M (4*32=128)
    const int wn   = wave >> 2;       // 2 waves along N (2*64=128)
    const int half = lane >> 4;
    const int lrow = lane & 15;

    // ---- async staging geometry: 8 x b128 per thread per tile (4 A + 4 B)
    const int crow = tid >> 3;            // 0..31 (row group)
    const int ccol = (tid & 7) * 8;       // half-offset within BK (8 halves = 16B)
    uint64_t gA[4], gB[4];
    uint32_t lA[4], lB[4];
    const uint32_t sA0 = (uint32_t)(uintptr_t)&sA[0][0];
    const uint32_t sB0 = (uint32_t)(uintptr_t)&sB[0][0];
    const uint32_t bufA = (uint32_t)(BM * LDKH * 2);   // bytes per A buffer
    const uint32_t bufB = (uint32_t)(BN * LDKH * 2);
#pragma unroll
    for (int t = 0; t < 4; ++t) {
        const int r = crow + 32 * t;
        gA[t] = (uint64_t)(uintptr_t)(A + (size_t)(m0 + r) * K_DIM + ccol);
        gB[t] = (uint64_t)(uintptr_t)(W + (size_t)(n0 + r) * K_DIM + ccol);
        lA[t] = sA0 + (uint32_t)((r * LDKH + ccol) * 2);
        lB[t] = sB0 + (uint32_t)((r * LDKH + ccol) * 2);
    }

    auto prefetch = [&](int i, int buf) {
        const uint64_t koff = (uint64_t)i * (BK * 2);   // bytes along K
        const uint32_t bo_a = (uint32_t)buf * bufA;
        const uint32_t bo_b = (uint32_t)buf * bufB;
#pragma unroll
        for (int t = 0; t < 4; ++t) async_ld_b128(lA[t] + bo_a, gA[t] + koff);
#pragma unroll
        for (int t = 0; t < 4; ++t) async_ld_b128(lB[t] + bo_b, gB[t] + koff);
    };

    v8f acc[2][4];
#pragma unroll
    for (int i = 0; i < 2; ++i)
#pragma unroll
        for (int j = 0; j < 4; ++j)
            acc[i][j] = (v8f){0.f, 0.f, 0.f, 0.f, 0.f, 0.f, 0.f, 0.f};

    auto compute = [&](int buf) {
#pragma unroll
        for (int s = 0; s < 2; ++s) {      // two 32-wide k-substeps
            v16h a[2], b[4];
#pragma unroll
            for (int i = 0; i < 2; ++i) {
                const int row = wm * 32 + i * 16 + lrow;
                const _Float16* p = &sA[buf][row * LDKH + s * 32];
                v8h lo = *(const v8h*)(p + half * 8);
                v8h hi = *(const v8h*)(p + 16 + half * 8);
                a[i] = __builtin_shufflevector(lo, hi, 0, 1, 2, 3, 4, 5, 6, 7,
                                                       8, 9, 10, 11, 12, 13, 14, 15);
            }
#pragma unroll
            for (int j = 0; j < 4; ++j) {
                const int brow = wn * 64 + j * 16 + lrow;
                const _Float16* p = &sB[buf][brow * LDKH + s * 32];
                v8h lo = *(const v8h*)(p + half * 8);
                v8h hi = *(const v8h*)(p + 16 + half * 8);
                b[j] = __builtin_shufflevector(lo, hi, 0, 1, 2, 3, 4, 5, 6, 7,
                                                       8, 9, 10, 11, 12, 13, 14, 15);
            }
#pragma unroll
            for (int i = 0; i < 2; ++i)
#pragma unroll
                for (int j = 0; j < 4; ++j)
                    acc[i][j] = __builtin_amdgcn_wmma_f32_16x16x32_f16(
                        false, a[i], false, b[j], (short)0, acc[i][j], false, false);
        }
    };

    // ---- software pipeline: 1-tile prefetch distance via ASYNCcnt
    prefetch(0, 0);
    int buf = 0;
    for (int i = 0; i < NK - 1; ++i) {
        prefetch(i + 1, buf ^ 1);      // writes other buffer (WAR-safe: barrier below)
        wait_async_le8();              // tile i (my 8 copies) landed; tile i+1 in flight
        __syncthreads();               // all waves' tile-i data visible
        compute(buf);                  // ds_load fragments consumed by wmma
        __syncthreads();               // all reads of buf done -> next iter may overwrite
        buf ^= 1;
    }
    wait_async_0();                    // last tile fully landed
    __syncthreads();
    compute(buf);

    // ---- epilogue: C frag rows r (lanes 0-15) / r+8 (lanes 16-31); NT stores
#pragma unroll
    for (int j = 0; j < 4; ++j) {
        const int col  = n0 + wn * 64 + j * 16 + lrow;
        const float bc = bias[col];
#pragma unroll
        for (int i = 0; i < 2; ++i) {
            const int rbase = m0 + wm * 32 + i * 16 + half * 8;
#pragma unroll
            for (int r = 0; r < 8; ++r)
                __builtin_nontemporal_store(acc[i][j][r] + bc,
                                            out + (size_t)(rbase + r) * N_DIM + col);
        }
    }
}

extern "C" void kernel_launch(void* const* d_in, const int* in_sizes, int n_in,
                              void* d_out, int out_size, void* d_ws, size_t ws_size,
                              hipStream_t stream) {
    (void)in_sizes; (void)n_in; (void)out_size; (void)ws_size;
    const float* x      = (const float*)d_in[0];
    const int*   qw     = (const int*)d_in[1];
    const float* scales = (const float*)d_in[2];
    const float* zeros  = (const float*)d_in[3];
    const float* bias   = (const float*)d_in[4];
    float*       out    = (float*)d_out;

    _Float16* a16 = (_Float16*)d_ws;                                       // 67.1 MB
    _Float16* w16 = (_Float16*)((char*)d_ws + (size_t)M_DIM * K_DIM * 2);  // 90.2 MB

    // pass 1: 8192*4096 / (256*4) blocks
    cvt_x_f16<<<(M_DIM * K_DIM) / (256 * 4), 256, 0, stream>>>(x, a16);
    // pass 2: grid (K/(256*4), N)
    dequant_w_f16<<<dim3(K_DIM / (256 * 4), N_DIM), 256, 0, stream>>>(qw, scales, zeros, w16);
    // pass 3: GEMM
    dim3 grid(N_DIM / BN, M_DIM / BM);   // 86 x 64
    exl2_wmma_gemm<<<grid, 256, 0, stream>>>(a16, w16, bias, out);
}
